// liif_out_80582176407647
// MI455X (gfx1250) — compile-verified, matrix operand order checked by reading, not run on hardware
//
#include <hip/hip_runtime.h>

// ---------------------------------------------------------------------------
// LIIF-style decode on MI455X (gfx1250).
// Batched-GEMM formulation: rows = B*tar_h*tar_w = 958464, MLP 258->128->64->64->9.
// ~88 GFLOP total, <100 MB true HBM traffic (feat fits in 192MB L2) => matrix-
// pipe bound => v_wmma_f32_16x16x32_f16 (f16 in, f32 accumulate).
// Each wave: 16 rows (M=16); A tile kept in registers (9 K-chunks of 32).
// Weights converted once to f16 WMMA-B fragments in d_ws; per 128-row block the
// 8 waves share weight chunks staged through LDS. Weight staging uses
// GLOBAL_LOAD_ASYNC_TO_LDS_B128 (ASYNCcnt) with double buffering when the
// toolchain exposes the builtin; otherwise a synchronous copy fallback.
// Output uses nontemporal stores (write-once stream, keep L2 for feat).
// ---------------------------------------------------------------------------

typedef __attribute__((ext_vector_type(16))) _Float16 v16h;
typedef __attribute__((ext_vector_type(8)))  float    v8f;
typedef int v4i_ __attribute__((vector_size(16)));   // matches builtin param type

#define BATCH 2
#define CCH   256
#define LHH   96
#define LWW   312
#define THH   384
#define TWW   1248
#define NPIX  (THH * TWW)        // 479232
#define ROWS  (BATCH * NPIX)     // 958464
#define H1    128
#define H2    64
#define H3    64
#define OUTD  9

// f16-element offsets of packed weight fragments in workspace
#define W1OFF 0
#define W1FR  (9 * 8)            // 9 k-chunks x 8 n-tiles
#define W2OFF (W1OFF + W1FR * 512)
#define W2FR  (4 * 4)
#define W3OFF (W2OFF + W2FR * 512)
#define W3FR  (2 * 4)
#define W4OFF (W3OFF + W3FR * 512)
#define W4FR  (2 * 1)
#define WTOT  (W4OFF + W4FR * 512)   // 50176 f16 = ~100 KB

#if defined(__AMDGCN__) && __has_builtin(__builtin_amdgcn_global_load_async_to_lds_b128)
#define ASYNC_STAGE 1
#else
#define ASYNC_STAGE 0
#endif

#if ASYNC_STAGE
// copy 32 bytes (16 f16) per thread, global -> LDS, async (2x b128)
__device__ __forceinline__ void stage32(const _Float16* g, _Float16* l) {
    __builtin_amdgcn_global_load_async_to_lds_b128(
        (__attribute__((address_space(1))) v4i_*)g,
        (__attribute__((address_space(3))) v4i_*)l, 0, 0);
    __builtin_amdgcn_global_load_async_to_lds_b128(
        (__attribute__((address_space(1))) v4i_*)(g + 8),
        (__attribute__((address_space(3))) v4i_*)(l + 8), 0, 0);
}
#if __has_builtin(__builtin_amdgcn_s_wait_asynccnt)
#define WAIT_ASYNC(n) __builtin_amdgcn_s_wait_asynccnt(n)
#else
#define WAIT_ASYNC(n) asm volatile("s_wait_asynccnt " #n ::: "memory")
#endif
#else
__device__ __forceinline__ void stage32(const _Float16* g, _Float16* l) {
    *(v16h*)l = *(const v16h*)g;   // global_load_b128 x2 + ds_store_b128 x2
}
#define WAIT_ASYNC(n) do {} while (0)
#endif

// K offset of the j-th f16 held by `lane` inside a 32-wide K chunk
// (CDNA5 16-bit A layout: lanes 0-15 hold K {0..7,16..23} as pairs,
//  lanes 16-31 hold K {8..15,24..31}; B uses the mirrored layout with N=lane%16)
__device__ __forceinline__ int kOf(int lane, int j) {
    int k = (j < 8) ? j : (j + 8);
    return k + ((lane & 16) ? 8 : 0);
}

// --------------------------- weight packing --------------------------------
__global__ void pack_weights(const float* __restrict__ W1, const float* __restrict__ W2,
                             const float* __restrict__ W3, const float* __restrict__ W4,
                             _Float16* __restrict__ ws) {
    int e = blockIdx.x * blockDim.x + threadIdx.x;
    if (e >= WTOT) return;
    const float* W; int base, NT, Kdim, Ndim, ldw;
    if (e < W2OFF)      { W = W1; base = W1OFF; NT = 8; Kdim = 258; Ndim = 128; ldw = 128; }
    else if (e < W3OFF) { W = W2; base = W2OFF; NT = 4; Kdim = 128; Ndim = 64;  ldw = 64;  }
    else if (e < W4OFF) { W = W3; base = W3OFF; NT = 4; Kdim = 64;  Ndim = 64;  ldw = 64;  }
    else                { W = W4; base = W4OFF; NT = 1; Kdim = 64;  Ndim = 9;   ldw = 9;   }
    int rel    = e - base;
    int frag   = rel >> 9;          // /512
    int within = rel & 511;
    int lane = within >> 4, j = within & 15;
    int kc = frag / NT, nt = frag % NT;
    int k = kc * 32 + kOf(lane, j);
    int n = nt * 16 + (lane & 15);
    float v = (k < Kdim && n < Ndim) ? W[k * ldw + n] : 0.0f;
    ws[e] = (_Float16)v;
}

// ------------------------------ main kernel --------------------------------
__global__ __launch_bounds__(256)
void liif_mlp(const float* __restrict__ feat, const _Float16* __restrict__ ws,
              const float* __restrict__ b1, const float* __restrict__ b2,
              const float* __restrict__ b3, const float* __restrict__ b4,
              float* __restrict__ out) {
    __shared__ __align__(32) _Float16 wbuf[8192];        // 16 KB: 2x4096 banks (L1) / flat (L2-L4)
    __shared__ __align__(32) _Float16 hst[8][16 * H1];   // 4 KB per-wave h stage

    const int tid  = threadIdx.x;
    const int wave = tid >> 5;
    const int lane = tid & 31;
    const int mrow = lane & 15;
    const long tile = (long)blockIdx.x * 8 + wave;
    const long g    = tile * 16 + mrow;                  // this lane's row

    // ---- prologue: kick off layer-1 weight chunk 0 into bank 0 ----
    stage32(&ws[W1OFF + tid * 16], &wbuf[tid * 16]);

    // ---- geometry: nearest-neighbor gather + relative coords ----
    int b  = (int)(g / NPIX);
    int p  = (int)(g - (long)b * NPIX);
    int ty = p / TWW, tx = p - ty * TWW;
    float gy = -1.0f + (2.0f * ty + 1.0f) / (float)THH;
    float gx = -1.0f + (2.0f * tx + 1.0f) / (float)TWW;
    const float eps = 1e-6f;
    float cy = fminf(fmaxf(gy, -1.0f + eps), 1.0f - eps);
    float cx = fminf(fmaxf(gx, -1.0f + eps), 1.0f - eps);
    int iy = (int)rintf(((cy + 1.0f) * LHH - 1.0f) * 0.5f);
    int ix = (int)rintf(((cx + 1.0f) * LWW - 1.0f) * 0.5f);
    iy = min(max(iy, 0), LHH - 1);
    ix = min(max(ix, 0), LWW - 1);
    float rel_y = (gy - (-1.0f + (2.0f * iy + 1.0f) / (float)LHH)) * (float)LHH;
    float rel_x = (gx - (-1.0f + (2.0f * ix + 1.0f) / (float)LWW)) * (float)LWW;
    const float* fb = feat + ((long)b * CCH * LHH + iy) * LWW + ix;  // + c*LHH*LWW

    // ---- build A fragments in registers (16x288 f16, A layout) ----
    v16h A[9];
#pragma unroll
    for (int c = 0; c < 8; ++c) {
        union { v16h v; _Float16 h[16]; } u;
#pragma unroll
        for (int j = 0; j < 16; ++j) {
            int k = c * 32 + kOf(lane, j);
            u.h[j] = (_Float16)fb[(long)k * (LHH * LWW)];
        }
        A[c] = u.v;
    }
    {   // chunk 8: K = 256..287 -> [rel_y, rel_x, 0...]
        union { v16h v; _Float16 h[16]; } u;
#pragma unroll
        for (int j = 0; j < 16; ++j) u.h[j] = (_Float16)0.0f;
        if (lane < 16) { u.h[0] = (_Float16)rel_y; u.h[1] = (_Float16)rel_x; }
        A[8] = u.v;
    }

    _Float16* hs = hst[wave];

    // ---------------- layer 1: (16x288) x (288x128), double-buffered --------
    v8f acc1[8];
    {
        v8f z = {};
#pragma unroll
        for (int nt = 0; nt < 8; ++nt) acc1[nt] = z;
    }
#pragma unroll
    for (int kc = 0; kc < 9; ++kc) {
        __syncthreads();                         // bank (kc+1)&1 no longer read
        if (kc < 8) {
            stage32(&ws[W1OFF + (kc + 1) * 4096 + tid * 16],
                    &wbuf[((kc + 1) & 1) * 4096 + tid * 16]);
            WAIT_ASYNC(2);                       // chunk kc landed (in-order)
        } else {
            WAIT_ASYNC(0);
        }
        __syncthreads();                         // chunk kc visible block-wide
        const _Float16* bank = &wbuf[(kc & 1) * 4096];
#pragma unroll
        for (int nt = 0; nt < 8; ++nt) {
            v16h bf = *(const v16h*)&bank[nt * 512 + lane * 16];
            acc1[nt] = __builtin_amdgcn_wmma_f32_16x16x32_f16(
                false, A[kc], false, bf, (short)0, acc1[nt], false, false);
        }
    }

    // issue layer-2 weight stage (8192 f16) now; overlap with epilogue math
    __syncthreads();                             // everyone done reading wbuf
    stage32(&ws[W2OFF + tid * 16],        &wbuf[tid * 16]);
    stage32(&ws[W2OFF + 4096 + tid * 16], &wbuf[4096 + tid * 16]);

    // bias + relu -> f16 h-stage [16][128]
#pragma unroll
    for (int nt = 0; nt < 8; ++nt) {
        int n = nt * 16 + (lane & 15);
        float bb = b1[n];
#pragma unroll
        for (int r = 0; r < 8; ++r) {
            float v = fmaxf(acc1[nt][r] + bb, 0.0f);
            int m = r + ((lane >> 4) << 3);
            hs[m * H1 + n] = (_Float16)v;
        }
    }
    asm volatile("s_wait_dscnt 0" ::: "memory");  // wave-local LDS RAW

    // ---------------- layer 2: (16x128) x (128x64) ----------------
    v16h A2[4];
#pragma unroll
    for (int c = 0; c < 4; ++c) {
        union { v16h v; _Float16 h[16]; } u;
#pragma unroll
        for (int j = 0; j < 16; ++j)
            u.h[j] = hs[(lane & 15) * H1 + c * 32 + kOf(lane, j)];
        A2[c] = u.v;
    }
    WAIT_ASYNC(0);
    __syncthreads();
    v8f acc2[4];
    { v8f z = {};
#pragma unroll
      for (int nt = 0; nt < 4; ++nt) acc2[nt] = z; }
#pragma unroll
    for (int kc = 0; kc < 4; ++kc) {
#pragma unroll
        for (int nt = 0; nt < 4; ++nt) {
            v16h bf = *(const v16h*)&wbuf[(kc * 4 + nt) * 512 + lane * 16];
            acc2[nt] = __builtin_amdgcn_wmma_f32_16x16x32_f16(
                false, A2[kc], false, bf, (short)0, acc2[nt], false, false);
        }
    }

    // issue layer-3 weight stage (4096 f16); overlap with epilogue math
    __syncthreads();
    stage32(&ws[W3OFF + tid * 16], &wbuf[tid * 16]);

#pragma unroll
    for (int nt = 0; nt < 4; ++nt) {
        int n = nt * 16 + (lane & 15);
        float bb = b2[n];
#pragma unroll
        for (int r = 0; r < 8; ++r) {
            float v = fmaxf(acc2[nt][r] + bb, 0.0f);
            int m = r + ((lane >> 4) << 3);
            hs[m * H2 + n] = (_Float16)v;
        }
    }
    asm volatile("s_wait_dscnt 0" ::: "memory");

    // ---------------- layer 3: (16x64) x (64x64) ----------------
    v16h A3[2];
#pragma unroll
    for (int c = 0; c < 2; ++c) {
        union { v16h v; _Float16 h[16]; } u;
#pragma unroll
        for (int j = 0; j < 16; ++j)
            u.h[j] = hs[(lane & 15) * H2 + c * 32 + kOf(lane, j)];
        A3[c] = u.v;
    }
    WAIT_ASYNC(0);
    __syncthreads();
    v8f acc3[4];
    { v8f z = {};
#pragma unroll
      for (int nt = 0; nt < 4; ++nt) acc3[nt] = z; }
#pragma unroll
    for (int kc = 0; kc < 2; ++kc) {
#pragma unroll
        for (int nt = 0; nt < 4; ++nt) {
            v16h bf = *(const v16h*)&wbuf[(kc * 4 + nt) * 512 + lane * 16];
            acc3[nt] = __builtin_amdgcn_wmma_f32_16x16x32_f16(
                false, A3[kc], false, bf, (short)0, acc3[nt], false, false);
        }
    }

    // issue layer-4 weight stage (1024 f16); overlap with epilogue math
    __syncthreads();
    if (tid < 64)
        stage32(&ws[W4OFF + tid * 16], &wbuf[tid * 16]);

#pragma unroll
    for (int nt = 0; nt < 4; ++nt) {
        int n = nt * 16 + (lane & 15);
        float bb = b3[n];
#pragma unroll
        for (int r = 0; r < 8; ++r) {
            float v = fmaxf(acc3[nt][r] + bb, 0.0f);
            int m = r + ((lane >> 4) << 3);
            hs[m * H3 + n] = (_Float16)v;
        }
    }
    asm volatile("s_wait_dscnt 0" ::: "memory");

    // ---------------- layer 4: (16x64) x (64x16) ----------------
    v16h A4[2];
#pragma unroll
    for (int c = 0; c < 2; ++c) {
        union { v16h v; _Float16 h[16]; } u;
#pragma unroll
        for (int j = 0; j < 16; ++j)
            u.h[j] = hs[(lane & 15) * H3 + c * 32 + kOf(lane, j)];
        A4[c] = u.v;
    }
    WAIT_ASYNC(0);
    __syncthreads();
    v8f acc4 = {};
#pragma unroll
    for (int kc = 0; kc < 2; ++kc) {
        v16h bf = *(const v16h*)&wbuf[kc * 512 + lane * 16];
        acc4 = __builtin_amdgcn_wmma_f32_16x16x32_f16(
            false, A4[kc], false, bf, (short)0, acc4, false, false);
    }

    // ---- write out (b, 9, th, tw); lane holds column n = lane%16 ----
    {
        int n = lane & 15;
        float bb = (n < OUTD) ? b4[n] : 0.0f;
        if (n < OUTD) {
#pragma unroll
            for (int r = 0; r < 8; ++r) {
                int m = r + ((lane >> 4) << 3);
                long gr = tile * 16 + m;
                int b_ = (int)(gr / NPIX);
                int p_ = (int)(gr - (long)b_ * NPIX);
                __builtin_nontemporal_store(acc4[r] + bb,
                    &out[((long)b_ * OUTD + n) * NPIX + p_]);
            }
        }
    }
}

// ------------------------------- launch ------------------------------------
extern "C" void kernel_launch(void* const* d_in, const int* in_sizes, int n_in,
                              void* d_out, int out_size, void* d_ws, size_t ws_size,
                              hipStream_t stream) {
    const float* feat = (const float*)d_in[0];
    const float* W1 = (const float*)d_in[1]; const float* b1 = (const float*)d_in[2];
    const float* W2 = (const float*)d_in[3]; const float* b2 = (const float*)d_in[4];
    const float* W3 = (const float*)d_in[5]; const float* b3 = (const float*)d_in[6];
    const float* W4 = (const float*)d_in[7]; const float* b4 = (const float*)d_in[8];
    _Float16* ws = (_Float16*)d_ws;

    pack_weights<<<(WTOT + 255) / 256, 256, 0, stream>>>(W1, W2, W3, W4, ws);

    // 958464 rows / (8 waves * 16 rows) = 7488 blocks, 256 threads (8 wave32)
    liif_mlp<<<ROWS / 128, 256, 0, stream>>>(feat, ws, b1, b2, b3, b4, (float*)d_out);
}
